// FrequencyDependentElectromagneticNeRF_13967233646665
// MI455X (gfx1250) — matrix operand order repaired and checked by reading, not compiled, not run on hardware
//
#include <hip/hip_runtime.h>
#include <math.h>

// ---------------- types ----------------
typedef __attribute__((ext_vector_type(8)))  _Float16 v8h;
typedef __attribute__((ext_vector_type(16))) _Float16 v16h;
typedef __attribute__((ext_vector_type(8)))  float    v8f;

union hv16 { v16h v; v8h h[2]; };

struct ResTab { int r[16]; };

// ---------------- constants ----------------
#define NLEV      16
#define LOG2T     19
#define TABLE_SZ  (1 << LOG2T)
#define TMASK     (TABLE_SZ - 1)
#define PM0       (73856093 % TABLE_SZ)
#define PM1       (19349663 % TABLE_SZ)
#define PM2       (83492791 % TABLE_SZ)
#define PI_F      3.14159265358979323846f

#define WAVES     4          // waves per block
#define BLOCK     (WAVES*32)
#define SPW       16         // samples per wave
#define SPB       (WAVES*SPW)
#define ACT_W     192        // activation row width (f16): h(128) | feat0(64)
#define HB_W      65         // head buffer row width (f32, bank-padded)

// dynamic LDS carve (bytes)
#define ACT_BYTES   (WAVES*SPW*ACT_W*2)        // 24576
#define HB_BYTES    (WAVES*SPW*HB_W*4)         // 16640
#define WSTAGE_F16  24576                      // max layer (W2p: 192x128)
#define WSTAGE_BYTES (WSTAGE_F16*2)            // 49152
#define SMEM_BYTES  (ACT_BYTES + HB_BYTES + WSTAGE_BYTES)  // 90368

// ---------------- weight pre-conversion to WMMA B-fragment layout ----------------
// Fragment f = ki*NT + t occupies 512 f16 (1 KB):
//   out[f*512 +        lane*8 + j]  = W[ki*32 + (lane/16)*16 + j     ][t*16 + lane%16]   (j=0..7)
//   out[f*512 + 256 +  lane*8 + j]  = W[ki*32 + (lane/16)*16 + 8 + j ][t*16 + lane%16]   (j=0..7)
// -> each lane's B fragment is two contiguous 16B chunks; across lanes each chunk
//    group spans 512B (2-phase, bank-conflict-free ds_load_b128).
__global__ void prep_w_kernel(const float* __restrict__ W, _Float16* __restrict__ out,
                              int fanIn, int Nout, int KI, int NT) {
  int tid = blockIdx.x * blockDim.x + threadIdx.x;
  int total = KI * NT * 32;
  if (tid >= total) return;
  int lane = tid & 31;
  int f    = tid >> 5;
  int ki   = f / NT;
  int t    = f - ki * NT;
  int half = lane >> 4;
  int n    = t * 16 + (lane & 15);
  _Float16* o0 = out + f * 512 + lane * 8;
  _Float16* o1 = o0 + 256;
#pragma unroll
  for (int j = 0; j < 8; ++j) {
    int k0 = ki * 32 + half * 16 + j;
    int k1 = k0 + 8;
    o0[j] = (_Float16)((k0 < fanIn) ? W[k0 * Nout + n] : 0.0f);
    o1[j] = (_Float16)((k1 < fanIn) ? W[k1 * Nout + n] : 0.0f);
  }
}

// ---------------- device helpers ----------------
__device__ __forceinline__ float sigm(float x) { return 1.0f / (1.0f + __expf(-x)); }

// cooperative copy of one layer's fragments into LDS (nf16 multiple of 8)
__device__ __forceinline__ void copy_w(_Float16* dst, const _Float16* __restrict__ src, int nf16) {
  int chunks = nf16 >> 3;
  for (int i = threadIdx.x; i < chunks; i += BLOCK)
    ((v8h*)dst)[i] = ((const v8h*)src)[i];
}

// GEMM over one wave's 16-sample tile: acc[t] += A(16xK) * B(Kx16*NT), B staged in LDS
template<int KI, int NT>
__device__ __forceinline__ void wave_gemm(const _Float16* actBase, int inOff,
                                          const _Float16* wb, v8f* acc) {
  const int lane = threadIdx.x & 31;
  const int m    = lane & 15;
  const int half = lane >> 4;
  const _Float16* rp = actBase + m * ACT_W + inOff + half * 16;
#pragma unroll
  for (int ki = 0; ki < KI; ++ki) {
    hv16 a;
    a.h[0] = *(const v8h*)(rp + ki * 32);
    a.h[1] = *(const v8h*)(rp + ki * 32 + 8);
#pragma unroll
    for (int t = 0; t < NT; ++t) {
      hv16 b;
      const _Float16* bp = wb + (ki * NT + t) * 512 + lane * 8;
      b.h[0] = *(const v8h*)(bp);
      b.h[1] = *(const v8h*)(bp + 256);
      acc[t] = __builtin_amdgcn_wmma_f32_16x16x32_f16(
          false, a.v, false, b.v, (short)0, acc[t], false, false);
    }
  }
}

template<int NT>
__device__ __forceinline__ void store_relu_f16(const v8f* acc, const float* __restrict__ bias,
                                               _Float16* actBase, int outOff) {
  const int lane = threadIdx.x & 31;
  const int m    = lane & 15;
  const int half = lane >> 4;
#pragma unroll
  for (int t = 0; t < NT; ++t) {
    float bb = bias[t * 16 + m];
#pragma unroll
    for (int r = 0; r < 8; ++r) {
      int row = half * 8 + r;
      float v = acc[t][r] + bb;
      actBase[row * ACT_W + outOff + t * 16 + m] = (_Float16)fmaxf(v, 0.0f);
    }
  }
}

template<int NT>
__device__ __forceinline__ void store_head(const v8f* acc, const float* __restrict__ bias,
                                           float* hb) {
  const int lane = threadIdx.x & 31;
  const int m    = lane & 15;
  const int half = lane >> 4;
#pragma unroll
  for (int t = 0; t < NT; ++t) {
    float bb = bias[t * 16 + m];
#pragma unroll
    for (int r = 0; r < 8; ++r) {
      hb[(half * 8 + r) * HB_W + t * 16 + m] = acc[t][r] + bb;
    }
  }
}

// ---------------- main fused kernel ----------------
__global__ void __launch_bounds__(BLOCK)
em_nerf_kernel(const float* __restrict__ xin, const float* __restrict__ freqin,
               const float* __restrict__ tables,
               const _Float16* __restrict__ W0p, const float* __restrict__ b0,
               const _Float16* __restrict__ W1p, const float* __restrict__ b1,
               const _Float16* __restrict__ W2p, const float* __restrict__ b2,
               const _Float16* __restrict__ W3p, const float* __restrict__ b3,
               const _Float16* __restrict__ ew1p, const float* __restrict__ eb1,
               const float* __restrict__ eg, const float* __restrict__ ebe,
               const float* __restrict__ ew2, const float* __restrict__ eb2,
               const _Float16* __restrict__ mw1p, const float* __restrict__ mb1,
               const float* __restrict__ mg, const float* __restrict__ mbe,
               const float* __restrict__ mw2, const float* __restrict__ mb2,
               const _Float16* __restrict__ pw1p, const float* __restrict__ pb1,
               const float* __restrict__ pw2, const float* __restrict__ pb2,
               const _Float16* __restrict__ matw1p, const float* __restrict__ matb1,
               const float* __restrict__ matw2, const float* __restrict__ matb2,
               const _Float16* __restrict__ sw1p, const float* __restrict__ sb1,
               const float* __restrict__ sw2, const float* __restrict__ sb2,
               float* __restrict__ out, ResTab rt) {
  extern __shared__ __align__(16) char smem[];
  _Float16* actAll = (_Float16*)smem;                       // [WAVES][16][ACT_W]
  float*    hbAll  = (float*)(smem + ACT_BYTES);            // [WAVES][16][HB_W]
  _Float16* wbuf   = (_Float16*)(smem + ACT_BYTES + HB_BYTES); // staged weights

  const int wave = threadIdx.x >> 5;
  const int lane = threadIdx.x & 31;
  const int base = blockIdx.x * SPB + wave * SPW;
  _Float16* actW = actAll + wave * (SPW * ACT_W);
  float*    hbW  = hbAll + wave * (SPW * HB_W);

  // ---------- phase 0: hash-grid encode -> act cols [128..159] ----------
#pragma unroll
  for (int it = 0; it < 8; ++it) {
    int task = it * 32 + lane;       // 16 samples x 16 levels
    int s = task >> 4;
    int l = task & 15;
    int gs = base + s;
    float px = fminf(fmaxf(xin[gs * 3 + 0], 0.0f), 1.0f);
    float py = fminf(fmaxf(xin[gs * 3 + 1], 0.0f), 1.0f);
    float pz = fminf(fmaxf(xin[gs * 3 + 2], 0.0f), 1.0f);
    int res = rt.r[l];
    float scf = (float)(res - 1);
    float sx = px * scf, sy = py * scf, sz = pz * scf;
    int fx = min(max((int)floorf(sx), 0), res - 1);
    int fy = min(max((int)floorf(sy), 0), res - 1);
    int fz = min(max((int)floorf(sz), 0), res - 1);
    int cx = min(fx + 1, res - 1);
    int cy = min(fy + 1, res - 1);
    int cz = min(fz + 1, res - 1);
    float wx = sx - (float)fx, wy = sy - (float)fy, wz = sz - (float)fz;
    float f0 = 0.0f, f1 = 0.0f;
    const float* tl = tables + (long)l * (TABLE_SZ * 2);
#pragma unroll
    for (int c = 0; c < 8; ++c) {
      int dx = (c >> 2) & 1, dy = (c >> 1) & 1, dz = c & 1;
      int ix = dx ? cx : fx, iy = dy ? cy : fy, iz = dz ? cz : fz;
      int h = ((ix * PM0) & TMASK) + ((iy * PM1) & TMASK) + ((iz * PM2) & TMASK);
      h &= TMASK;
      float cw = (dx ? wx : 1.0f - wx) * (dy ? wy : 1.0f - wy) * (dz ? wz : 1.0f - wz);
      const float* fp = tl + 2 * h;
      f0 += fp[0] * cw;
      f1 += fp[1] * cw;
    }
    actW[s * ACT_W + 128 + 2 * l]     = (_Float16)f0;
    actW[s * ACT_W + 128 + 2 * l + 1] = (_Float16)f1;
  }

  // ---------- phase 1: frequency encode -> act cols [160..179], pad [180..191] ----------
#pragma unroll
  for (int it = 0; it < 5; ++it) {
    int t = it * 32 + lane;          // 16 samples x 10 bands
    if (t < 160) {
      int s = t / 10;
      int bnd = t - s * 10;
      float f = fminf(fmaxf(freqin[base + s], 1e6f), 1e12f);
      float nf = (log10f(f) - 6.0f) / 6.0f;
      float a = (float)(1 << bnd) * PI_F * nf;
      actW[s * ACT_W + 160 + 2 * bnd]     = (_Float16)sinf(a);
      actW[s * ACT_W + 160 + 2 * bnd + 1] = (_Float16)cosf(a);
    }
  }
  if (lane < 16) {
#pragma unroll
    for (int c = 0; c < 12; ++c) actW[lane * ACT_W + 180 + c] = (_Float16)0.0f;
  }

  // stage = sync (everyone done with previous wbuf) ; copy ; sync (wbuf ready)
#define STAGE_W(SRC, NF16) do { __syncthreads(); copy_w(wbuf, (SRC), (NF16)); __syncthreads(); } while (0)

  // ---------- main MLP ----------
  const v8f vzero = {0.f, 0.f, 0.f, 0.f, 0.f, 0.f, 0.f, 0.f};
  v8f acc[8];

  // L0: feat0(64) @ W0 -> h0 (cols 0..127)
  STAGE_W(W0p, 8192);
#pragma unroll
  for (int t = 0; t < 8; ++t) acc[t] = vzero;
  wave_gemm<2, 8>(actW, 128, wbuf, acc);
  store_relu_f16<8>(acc, b0, actW, 0);

  // L1: h0 @ W1 -> h1
  STAGE_W(W1p, 16384);
#pragma unroll
  for (int t = 0; t < 8; ++t) acc[t] = vzero;
  wave_gemm<4, 8>(actW, 0, wbuf, acc);
  store_relu_f16<8>(acc, b1, actW, 0);

  // L2: [h1 | feat0](192) @ W2 -> h2 (skip concat is free: feat0 lives in cols 128..191)
  STAGE_W(W2p, 24576);
#pragma unroll
  for (int t = 0; t < 8; ++t) acc[t] = vzero;
  wave_gemm<6, 8>(actW, 0, wbuf, acc);
  store_relu_f16<8>(acc, b2, actW, 0);

  // L3: h2 @ W3 -> h3
  STAGE_W(W3p, 16384);
#pragma unroll
  for (int t = 0; t < 8; ++t) acc[t] = vzero;
  wave_gemm<4, 8>(actW, 0, wbuf, acc);
  store_relu_f16<8>(acc, b3, actW, 0);

  // ---------- heads ----------
  // epilogues use all 32 lanes: sample = lane%16, half-range = (lane/16)*half, combine shfl_xor(16)
  float Ea[3] = {0, 0, 0}, Ba[3] = {0, 0, 0}, ph[6] = {0, 0, 0, 0, 0, 0};
  float epsv = 0, muv = 0, condv = 0, sv[4] = {0, 0, 0, 0};
  const float* myrow = hbW + (lane & 15) * HB_W;
  const int h32 = (lane >> 4) * 32;   // for 64-wide heads
  const int h16 = (lane >> 4) * 16;   // for 32-wide head

  // E head: LN -> relu -> 64x3 -> tanh
  {
    STAGE_W(ew1p, 8192);
#pragma unroll
    for (int t = 0; t < 4; ++t) acc[t] = vzero;
    wave_gemm<4, 4>(actW, 0, wbuf, acc);
    store_head<4>(acc, eb1, hbW);
    float s1 = 0.f, s2 = 0.f;
    for (int n = 0; n < 32; ++n) { float v = myrow[h32 + n]; s1 += v; s2 += v * v; }
    s1 += __shfl_xor(s1, 16);
    s2 += __shfl_xor(s2, 16);
    float mean = s1 * (1.0f / 64.0f);
    float var  = s2 * (1.0f / 64.0f) - mean * mean;
    float inv  = rsqrtf(var + 1e-5f);
    float d0 = 0.f, d1 = 0.f, d2 = 0.f;
    for (int n = 0; n < 32; ++n) {
      int nn = h32 + n;
      float tv = fmaxf((myrow[nn] - mean) * inv * eg[nn] + ebe[nn], 0.0f);
      d0 += tv * ew2[nn * 3 + 0];
      d1 += tv * ew2[nn * 3 + 1];
      d2 += tv * ew2[nn * 3 + 2];
    }
    d0 += __shfl_xor(d0, 16);
    d1 += __shfl_xor(d1, 16);
    d2 += __shfl_xor(d2, 16);
    Ea[0] = tanhf(d0 + eb2[0]);
    Ea[1] = tanhf(d1 + eb2[1]);
    Ea[2] = tanhf(d2 + eb2[2]);
  }

  // B head
  {
    STAGE_W(mw1p, 8192);
#pragma unroll
    for (int t = 0; t < 4; ++t) acc[t] = vzero;
    wave_gemm<4, 4>(actW, 0, wbuf, acc);
    store_head<4>(acc, mb1, hbW);
    float s1 = 0.f, s2 = 0.f;
    for (int n = 0; n < 32; ++n) { float v = myrow[h32 + n]; s1 += v; s2 += v * v; }
    s1 += __shfl_xor(s1, 16);
    s2 += __shfl_xor(s2, 16);
    float mean = s1 * (1.0f / 64.0f);
    float var  = s2 * (1.0f / 64.0f) - mean * mean;
    float inv  = rsqrtf(var + 1e-5f);
    float d0 = 0.f, d1 = 0.f, d2 = 0.f;
    for (int n = 0; n < 32; ++n) {
      int nn = h32 + n;
      float tv = fmaxf((myrow[nn] - mean) * inv * mg[nn] + mbe[nn], 0.0f);
      d0 += tv * mw2[nn * 3 + 0];
      d1 += tv * mw2[nn * 3 + 1];
      d2 += tv * mw2[nn * 3 + 2];
    }
    d0 += __shfl_xor(d0, 16);
    d1 += __shfl_xor(d1, 16);
    d2 += __shfl_xor(d2, 16);
    Ba[0] = tanhf(d0 + mb2[0]);
    Ba[1] = tanhf(d1 + mb2[1]);
    Ba[2] = tanhf(d2 + mb2[2]);
  }

  // phase head: relu -> 32x6
  {
    STAGE_W(pw1p, 4096);
#pragma unroll
    for (int t = 0; t < 2; ++t) acc[t] = vzero;
    wave_gemm<4, 2>(actW, 0, wbuf, acc);
    store_head<2>(acc, pb1, hbW);
    float d[6] = {0, 0, 0, 0, 0, 0};
    for (int n = 0; n < 16; ++n) {
      int nn = h16 + n;
      float tv = fmaxf(myrow[nn], 0.0f);
#pragma unroll
      for (int j = 0; j < 6; ++j) d[j] += tv * pw2[nn * 6 + j];
    }
#pragma unroll
    for (int j = 0; j < 6; ++j) {
      d[j] += __shfl_xor(d[j], 16);
      ph[j] = d[j] + pb2[j];
    }
  }

  // material head: relu -> 64x3 -> sigmoids
  {
    STAGE_W(matw1p, 8192);
#pragma unroll
    for (int t = 0; t < 4; ++t) acc[t] = vzero;
    wave_gemm<4, 4>(actW, 0, wbuf, acc);
    store_head<4>(acc, matb1, hbW);
    float d0 = 0.f, d1 = 0.f, d2 = 0.f;
    for (int n = 0; n < 32; ++n) {
      int nn = h32 + n;
      float tv = fmaxf(myrow[nn], 0.0f);
      d0 += tv * matw2[nn * 3 + 0];
      d1 += tv * matw2[nn * 3 + 1];
      d2 += tv * matw2[nn * 3 + 2];
    }
    d0 += __shfl_xor(d0, 16);
    d1 += __shfl_xor(d1, 16);
    d2 += __shfl_xor(d2, 16);
    epsv  = 1.0f + sigm(d0 + matb2[0]) * 10.0f;
    muv   = 1.0f + sigm(d1 + matb2[1]) * 2.0f;
    condv = sigm(d2 + matb2[2]) * 0.01f;
  }

  // source head: relu -> 64x4
  {
    STAGE_W(sw1p, 8192);
#pragma unroll
    for (int t = 0; t < 4; ++t) acc[t] = vzero;
    wave_gemm<4, 4>(actW, 0, wbuf, acc);
    store_head<4>(acc, sb1, hbW);
    float d[4] = {0, 0, 0, 0};
    for (int n = 0; n < 32; ++n) {
      int nn = h32 + n;
      float tv = fmaxf(myrow[nn], 0.0f);
#pragma unroll
      for (int j = 0; j < 4; ++j) d[j] += tv * sw2[nn * 4 + j];
    }
#pragma unroll
    for (int j = 0; j < 4; ++j) {
      d[j] += __shfl_xor(d[j], 16);
      sv[j] = d[j] + sb2[j];
    }
  }

  // ---------- final packing: E(3) B(3) eps mu cond charge current(3) ----------
  // TIME == 0  =>  cos(omega*TIME + phase) == cos(phase)
  if (lane < 16) {
    int gs = base + lane;
    float* o = out + (long)gs * 13;
    o[0]  = Ea[0] * cosf(ph[0]);
    o[1]  = Ea[1] * cosf(ph[1]);
    o[2]  = Ea[2] * cosf(ph[2]);
    o[3]  = Ba[0] * cosf(ph[3]);
    o[4]  = Ba[1] * cosf(ph[4]);
    o[5]  = Ba[2] * cosf(ph[5]);
    o[6]  = epsv;
    o[7]  = muv;
    o[8]  = condv;
    o[9]  = sv[0];
    o[10] = sv[1];
    o[11] = sv[2];
    o[12] = sv[3];
  }
#undef STAGE_W
}

// ---------------- host launcher ----------------
extern "C" void kernel_launch(void* const* d_in, const int* in_sizes, int n_in,
                              void* d_out, int out_size, void* d_ws, size_t ws_size,
                              hipStream_t stream) {
  const float* x     = (const float*)d_in[0];
  const float* freq  = (const float*)d_in[1];
  const float* tabs  = (const float*)d_in[2];
  const float* W0  = (const float*)d_in[3];   const float* b0  = (const float*)d_in[4];
  const float* W1  = (const float*)d_in[5];   const float* b1  = (const float*)d_in[6];
  const float* W2  = (const float*)d_in[7];   const float* b2  = (const float*)d_in[8];
  const float* W3  = (const float*)d_in[9];   const float* b3  = (const float*)d_in[10];
  const float* ew1 = (const float*)d_in[11];  const float* eb1 = (const float*)d_in[12];
  const float* ew2 = (const float*)d_in[13];  const float* eb2 = (const float*)d_in[14];
  const float* mw1 = (const float*)d_in[15];  const float* mb1 = (const float*)d_in[16];
  const float* mw2 = (const float*)d_in[17];  const float* mb2 = (const float*)d_in[18];
  const float* pw1 = (const float*)d_in[19];  const float* pb1 = (const float*)d_in[20];
  const float* pw2 = (const float*)d_in[21];  const float* pb2 = (const float*)d_in[22];
  const float* mtw1 = (const float*)d_in[23]; const float* mtb1 = (const float*)d_in[24];
  const float* mtw2 = (const float*)d_in[25]; const float* mtb2 = (const float*)d_in[26];
  const float* sw1 = (const float*)d_in[27];  const float* sb1 = (const float*)d_in[28];
  const float* sw2 = (const float*)d_in[29];  const float* sb2 = (const float*)d_in[30];
  const float* eg  = (const float*)d_in[31];  const float* ebe = (const float*)d_in[32];
  const float* mg  = (const float*)d_in[33];  const float* mbe = (const float*)d_in[34];
  float* out = (float*)d_out;

  // resolution table, same double-precision chain as the reference
  ResTab rt;
  {
    double Bv = exp((log(512.0) - log(16.0)) / 15.0);
    for (int l = 0; l < 16; ++l) {
      int r = (int)(16.0 * pow(Bv, (double)l));
      rt.r[l] = r < 512 ? r : 512;
    }
  }

  // workspace layout (f16 elements), 32B-aligned chunks
  _Float16* ws = (_Float16*)d_ws;
  _Float16* W0p   = ws;            // 64*128   = 8192
  _Float16* W1p   = W0p + 8192;    // 128*128  = 16384
  _Float16* W2p   = W1p + 16384;   // 192*128  = 24576
  _Float16* W3p   = W2p + 24576;   // 128*128  = 16384
  _Float16* ew1p  = W3p + 16384;   // 128*64   = 8192
  _Float16* mw1p  = ew1p + 8192;   // 8192
  _Float16* pw1p  = mw1p + 8192;   // 128*32   = 4096
  _Float16* mtw1p = pw1p + 4096;   // 8192
  _Float16* sw1p  = mtw1p + 8192;  // 8192
  (void)ws_size; (void)n_in; (void)out_size;

  auto prep = [&](const float* W, _Float16* o, int fanIn, int Nout, int KI, int NT) {
    int total = KI * NT * 32;
    int blocks = (total + 255) / 256;
    prep_w_kernel<<<blocks, 256, 0, stream>>>(W, o, fanIn, Nout, KI, NT);
  };
  prep(W0,  W0p,   52, 128, 2, 8);
  prep(W1,  W1p,  128, 128, 4, 8);
  prep(W2,  W2p,  180, 128, 6, 8);
  prep(W3,  W3p,  128, 128, 4, 8);
  prep(ew1, ew1p, 128,  64, 4, 4);
  prep(mw1, mw1p, 128,  64, 4, 4);
  prep(pw1, pw1p, 128,  32, 4, 2);
  prep(mtw1, mtw1p, 128, 64, 4, 4);
  prep(sw1, sw1p, 128,  64, 4, 4);

  int N = in_sizes[0] / 3;          // 262144
  int grid = N / SPB;               // 4096 blocks of 128 threads (exact)
  em_nerf_kernel<<<grid, BLOCK, SMEM_BYTES, stream>>>(
      x, freq, tabs,
      W0p, b0, W1p, b1, W2p, b2, W3p, b3,
      ew1p, eb1, eg, ebe, ew2, eb2,
      mw1p, mb1, mg, mbe, mw2, mb2,
      pw1p, pb1, pw2, pb2,
      mtw1p, mtb1, mtw2, mtb2,
      sw1p, sb1, sw2, sb2,
      out, rt);
}
